// EncoderBlock_89498528514071
// MI455X (gfx1250) — matrix operand order repaired and checked by reading, not compiled, not run on hardware
//
#include <hip/hip_runtime.h>
#include <hip/hip_bf16.h>
#include <math.h>

// Problem constants (match reference)
#define BB   2
#define SS   2048
#define DD   1024
#define HH   16
#define HDD  64
#define MLPD 4096
#define EPSF 1e-6f

typedef __bf16 bf16_t;
typedef __attribute__((ext_vector_type(16))) __bf16 v16bf;
typedef __attribute__((ext_vector_type(8)))  __bf16 v8bf;
typedef __attribute__((ext_vector_type(8)))  float  v8f;

#if __has_builtin(__builtin_amdgcn_global_load_async_to_lds_b128) && \
    __has_builtin(__builtin_amdgcn_s_wait_asynccnt)
#define USE_ASYNC 1
// Builtin signature (from hipcc diagnostic): param0 = v4i in AS1 (__device__),
// param1 = v4i in AS3 (__shared__), then imm offset + cpol.
typedef int v4i_t __attribute__((vector_size(16)));
typedef __attribute__((address_space(1))) v4i_t gv4i;
typedef __attribute__((address_space(3))) v4i_t lv4i;
#else
#define USE_ASYNC 0
#endif

static __device__ __forceinline__ v8bf ld8(const bf16_t* p) { return *(const v8bf*)p; }

static __device__ __forceinline__ v16bf cat8(v8bf lo, v8bf hi) {
  v16bf r;
#pragma unroll
  for (int i = 0; i < 8; ++i) { r[i] = lo[i]; r[i + 8] = hi[i]; }
  return r;
}

static __device__ __forceinline__ v8f wmma_bf(v16bf a, v16bf b, v8f c) {
  // D = A(16x32 bf16) x B(32x16 bf16) + C(16x16 f32)
  return __builtin_amdgcn_wmma_f32_16x16x32_bf16(false, a, false, b, (short)0, c,
                                                 false, false);
}

// ---------------------------------------------------------------------------
// Weight convert + transpose: W[K][N] fp32 -> Wt[N][K] bf16
// ---------------------------------------------------------------------------
__global__ __launch_bounds__(256) void cvt_kernel(const float* __restrict__ w,
                                                  bf16_t* __restrict__ wt,
                                                  int K, int N) {
  int idx = blockIdx.x * 256 + threadIdx.x;
  if (idx < K * N) {
    int n = idx / K;
    int k = idx - n * K;
    wt[idx] = (bf16_t)w[(size_t)k * N + n];
  }
}

// ---------------------------------------------------------------------------
// LayerNorm rows of [rows x DD] fp32 -> bf16
// ---------------------------------------------------------------------------
__global__ __launch_bounds__(256) void ln_kernel(const float* __restrict__ x,
                                                 const float* __restrict__ g,
                                                 const float* __restrict__ bta,
                                                 bf16_t* __restrict__ out) {
  __shared__ float sh1[256];
  __shared__ float sh2[256];
  const int row = blockIdx.x;
  const float* xr = x + (size_t)row * DD;
  const int t = threadIdx.x;
  float s = 0.f, s2 = 0.f;
  for (int i = t; i < DD; i += 256) { float v = xr[i]; s += v; s2 += v * v; }
  sh1[t] = s; sh2[t] = s2;
  __syncthreads();
  for (int st = 128; st > 0; st >>= 1) {
    if (t < st) { sh1[t] += sh1[t + st]; sh2[t] += sh2[t + st]; }
    __syncthreads();
  }
  const float mean = sh1[0] * (1.0f / DD);
  const float var  = sh2[0] * (1.0f / DD) - mean * mean;
  const float rstd = rsqrtf(var + EPSF);
  for (int i = t; i < DD; i += 256)
    out[(size_t)row * DD + i] = (bf16_t)((xr[i] - mean) * rstd * g[i] + bta[i]);
}

// ---------------------------------------------------------------------------
// bf16 WMMA GEMM: C[MxN] = A[MxK] * Bt[NxK]^T, fused epilogues.
//   mode 0: out_bf = q/k scatter to [B,H,S,HD], +bias
//   mode 1: out_bf = v scatter to [B,H,HD,S] (transposed), +bias
//   mode 2: out_f  = resid + acc + bias        (row-major fp32)
//   mode 3: out_bf = gelu(acc + bias)          (row-major bf16, exact erf)
// Block = 256 threads = 8 waves, each wave a 32(M)x64(N) tile -> block 256x64.
// B tile (shared by all 8 waves) staged through double-buffered LDS, copied
// with gfx1250 async-to-LDS when available; K-loop software pipelined.
// ---------------------------------------------------------------------------
#define LDS_STRIDE 40  // 32 bf16 + 8 pad (80B rows) to spread LDS banks
__global__ __launch_bounds__(256) void gemm_bf16_kernel(
    const bf16_t* __restrict__ A, const bf16_t* __restrict__ Bt,
    const float* __restrict__ bias, const float* __restrict__ resid,
    float* __restrict__ outF, bf16_t* __restrict__ outB,
    int M, int N, int K, int mode) {
  __shared__ alignas(16) bf16_t ldsB[2][64 * LDS_STRIDE];

  const int tid   = threadIdx.x;
  const int wave  = tid >> 5;
  const int lane  = tid & 31;
  const int colL  = lane & 15;
  const int lg    = lane >> 4;
  const int rbase = (lane & 16) ? 8 : 0;
  const int m0 = blockIdx.y * 256 + wave * 32;
  const int n0 = blockIdx.x * 64;

  // B-tile staging: 64 cols x 32 K x bf16 = 4KB/stage; 256 thr x 16B each.
  const int sc = tid >> 2;        // column 0..63
  const int sq = (tid & 3) * 8;   // k-offset 0,8,16,24
  const bf16_t* gB = Bt + (size_t)(n0 + sc) * K + sq;
  bf16_t* lB[2] = { &ldsB[0][sc * LDS_STRIDE + sq], &ldsB[1][sc * LDS_STRIDE + sq] };

  // Stage 0
#if USE_ASYNC
  __builtin_amdgcn_global_load_async_to_lds_b128((gv4i*)gB, (lv4i*)lB[0], 0, 0);
  __builtin_amdgcn_s_wait_asynccnt(0);
  __syncthreads();
#else
  { v8bf st0 = ld8(gB); *(v8bf*)lB[0] = st0; }
  __syncthreads();
#endif

  const bf16_t* ap0 = A + (size_t)(m0 + colL) * K;
  const bf16_t* ap1 = A + (size_t)(m0 + 16 + colL) * K;

  v8f acc[8] = {};

  // Preload A fragments for kk = 0
  v16bf a0 = cat8(ld8(ap0 + lg * 8), ld8(ap0 + 16 + lg * 8));
  v16bf a1 = cat8(ld8(ap1 + lg * 8), ld8(ap1 + 16 + lg * 8));

  for (int kk = 0; kk < K; kk += 32) {
    const int buf  = (kk >> 5) & 1;
    const bool more = (kk + 32) < K;

    // ---- issue next B stage + next A fragments (overlap with WMMAs) ------
#if !USE_ASYNC
    v8bf nxtB;
#endif
    if (more) {
#if USE_ASYNC
      __builtin_amdgcn_global_load_async_to_lds_b128((gv4i*)(gB + kk + 32),
                                                     (lv4i*)lB[buf ^ 1], 0, 0);
#else
      nxtB = ld8(gB + kk + 32);
#endif
    }
    v16bf na0, na1;
    if (more) {
      na0 = cat8(ld8(ap0 + kk + 32 + lg * 8), ld8(ap0 + kk + 48 + lg * 8));
      na1 = cat8(ld8(ap1 + kk + 32 + lg * 8), ld8(ap1 + kk + 48 + lg * 8));
    }

    // ---- consume current B stage from LDS --------------------------------
    const bf16_t* lb = ldsB[buf];
#pragma unroll
    for (int j = 0; j < 4; ++j) {
      const bf16_t* bp = lb + (j * 16 + colL) * LDS_STRIDE + lg * 8;
      v16bf b = cat8(ld8(bp), ld8(bp + 16));
      acc[j]     = wmma_bf(a0, b, acc[j]);
      acc[4 + j] = wmma_bf(a1, b, acc[4 + j]);
    }

    if (more) {
#if USE_ASYNC
      __builtin_amdgcn_s_wait_asynccnt(0);
      __syncthreads();
#else
      __syncthreads();
      *(v8bf*)lB[buf ^ 1] = nxtB;
      __syncthreads();
#endif
      a0 = na0;
      a1 = na1;
    }
  }

  // ---- epilogue ----------------------------------------------------------
#pragma unroll
  for (int i = 0; i < 2; ++i) {
#pragma unroll
    for (int j = 0; j < 4; ++j) {
      const int n = n0 + j * 16 + colL;
      const float bv = bias[n];
#pragma unroll
      for (int r = 0; r < 8; ++r) {
        const int m = m0 + i * 16 + r + rbase;
        const float val = acc[i * 4 + j][r] + bv;
        if (mode == 0) {
          int b = m >> 11, s = m & (SS - 1), h = n >> 6, hd = n & 63;
          outB[((size_t)(b * HH + h) * SS + s) * HDD + hd] = (bf16_t)val;
        } else if (mode == 1) {
          int b = m >> 11, s = m & (SS - 1), h = n >> 6, hd = n & 63;
          outB[((size_t)(b * HH + h) * HDD + hd) * SS + s] = (bf16_t)val;
        } else if (mode == 2) {
          size_t idx = (size_t)m * N + n;
          outF[idx] = resid[idx] + val;
        } else {
          float gl = 0.5f * val * (1.0f + erff(val * 0.70710678118654752f));
          outB[(size_t)m * N + n] = (bf16_t)gl;
        }
      }
    }
  }
}

// ---------------------------------------------------------------------------
// Flash attention. 4 waves/block, 16 queries/wave. Scores computed TRANSPOSED
// (S^T = K * Q^T) so the C-fragment (lane=query, VGPR=key) maps directly onto
// the 16x32 bf16 A-fragment of the P*V WMMA: zero cross-lane shuffles between
// softmax and PV. K-stream register-pipelined: next tile's K fragments are in
// flight while softmax VALU work runs.
// Q,K: [B,H,S,HD] bf16.  Vt: [B,H,HD,S] bf16.  Out: [B,S,D] bf16.
// ---------------------------------------------------------------------------
__global__ __launch_bounds__(128) void attn_kernel(
    const bf16_t* __restrict__ Q, const bf16_t* __restrict__ Kc,
    const bf16_t* __restrict__ Vt, bf16_t* __restrict__ Oa) {
  const int wave  = threadIdx.x >> 5;
  const int lane  = threadIdx.x & 31;
  const int colL  = lane & 15;
  const int lg    = lane >> 4;
  const int rbase = (lane & 16) ? 8 : 0;
  const int bh = blockIdx.y;  // b*H + h
  const int b  = bh / HH;
  const int h  = bh % HH;
  const int q0 = blockIdx.x * 64 + wave * 16;

  const bf16_t* qp = Q  + (size_t)bh * SS * HDD;
  const bf16_t* kp = Kc + (size_t)bh * SS * HDD;
  const bf16_t* vp = Vt + (size_t)bh * HDD * SS;

  // Resident B-fragments of Q^T (hd chunks [0,32) and [32,64)).
  const bf16_t* qrow = qp + (size_t)(q0 + colL) * HDD;
  const v16bf bq0 = cat8(ld8(qrow + lg * 8),      ld8(qrow + 16 + lg * 8));
  const v16bf bq1 = cat8(ld8(qrow + 32 + lg * 8), ld8(qrow + 48 + lg * 8));

  v8f o[4] = {};
  float mrow = -INFINITY, lrow = 0.0f;
  const float scale = 0.125f;  // HD^-0.5

  // Preload K fragments for tile kt = 0.
  const bf16_t* k0p = kp + (size_t)colL * HDD;
  const bf16_t* k1p = kp + (size_t)(16 + colL) * HDD;
  v16bf a0lo = cat8(ld8(k0p + lg * 8),      ld8(k0p + 16 + lg * 8));
  v16bf a0hi = cat8(ld8(k0p + 32 + lg * 8), ld8(k0p + 48 + lg * 8));
  v16bf a1lo = cat8(ld8(k1p + lg * 8),      ld8(k1p + 16 + lg * 8));
  v16bf a1hi = cat8(ld8(k1p + 32 + lg * 8), ld8(k1p + 48 + lg * 8));

  for (int kt = 0; kt < SS; kt += 32) {
    // ---- S^T tiles for 32 keys -------------------------------------------
    v8f c0 = {}, c1 = {};
    c0 = wmma_bf(a0lo, bq0, c0);  c0 = wmma_bf(a0hi, bq1, c0);
    c1 = wmma_bf(a1lo, bq0, c1);  c1 = wmma_bf(a1hi, bq1, c1);

    // Issue next tile's K fragments now; softmax below hides their latency.
    if (kt + 32 < SS) {
      const bf16_t* nk0 = kp + (size_t)(kt + 32 + colL) * HDD;
      const bf16_t* nk1 = kp + (size_t)(kt + 48 + colL) * HDD;
      a0lo = cat8(ld8(nk0 + lg * 8),      ld8(nk0 + 16 + lg * 8));
      a0hi = cat8(ld8(nk0 + 32 + lg * 8), ld8(nk0 + 48 + lg * 8));
      a1lo = cat8(ld8(nk1 + lg * 8),      ld8(nk1 + 16 + lg * 8));
      a1hi = cat8(ld8(nk1 + 32 + lg * 8), ld8(nk1 + 48 + lg * 8));
    }

    // ---- online softmax (lane pair colL, colL+16 covers all 32 keys) -----
    float s0[8], s1[8];
    float mt = -INFINITY;
#pragma unroll
    for (int r = 0; r < 8; ++r) {
      s0[r] = c0[r] * scale;
      s1[r] = c1[r] * scale;
      mt = fmaxf(mt, fmaxf(s0[r], s1[r]));
    }
    mt = fmaxf(mt, __shfl_xor(mt, 16, 32));
    const float mnew = fmaxf(mrow, mt);
    const float corr = __expf(mrow - mnew);
    mrow = mnew;

    float lt = 0.0f;
    v16bf pA;  // A-fragment of P, built in place
#pragma unroll
    for (int r = 0; r < 8; ++r) {
      float p0 = __expf(s0[r] - mnew);
      float p1 = __expf(s1[r] - mnew);
      lt += p0 + p1;
      pA[r]     = (bf16_t)p0;
      pA[r + 8] = (bf16_t)p1;
    }
    lt += __shfl_xor(lt, 16, 32);
    lrow = lrow * corr + lt;

    // rescale output accumulators (per-query factor broadcast via shfl)
#pragma unroll
    for (int r = 0; r < 8; ++r) {
      const float f = __shfl(corr, r + rbase, 32);
      o[0][r] *= f; o[1][r] *= f; o[2][r] *= f; o[3][r] *= f;
    }

    // ---- P x V: B-fragments from Vt rows (contiguous over keys) ----------
#pragma unroll
    for (int j = 0; j < 4; ++j) {
      const bf16_t* vr = vp + (size_t)(j * 16 + colL) * SS + kt + lg * 8;
      v16bf bv = cat8(ld8(vr), ld8(vr + 16));
      o[j] = wmma_bf(pA, bv, o[j]);
    }
  }

  const float inv = 1.0f / lrow;
#pragma unroll
  for (int j = 0; j < 4; ++j) {
#pragma unroll
    for (int r = 0; r < 8; ++r) {
      const int m = q0 + r + rbase;
      const float f = __shfl(inv, r + rbase, 32);
      Oa[((size_t)(b * SS + m)) * DD + h * HDD + j * 16 + colL] =
          (bf16_t)(o[j][r] * f);
    }
  }
}

// ---------------------------------------------------------------------------
// Host-side orchestration
// ---------------------------------------------------------------------------
extern "C" void kernel_launch(void* const* d_in, const int* in_sizes, int n_in,
                              void* d_out, int out_size, void* d_ws,
                              size_t ws_size, hipStream_t stream) {
  const float* x    = (const float*)d_in[0];
  const float* ln1g = (const float*)d_in[1];
  const float* ln1b = (const float*)d_in[2];
  const float* wq   = (const float*)d_in[3];
  const float* bq   = (const float*)d_in[4];
  const float* wk   = (const float*)d_in[5];
  const float* bk   = (const float*)d_in[6];
  const float* wv   = (const float*)d_in[7];
  const float* bv   = (const float*)d_in[8];
  const float* wo   = (const float*)d_in[9];
  const float* bo   = (const float*)d_in[10];
  const float* ln2g = (const float*)d_in[11];
  const float* ln2b = (const float*)d_in[12];
  const float* w1   = (const float*)d_in[13];
  const float* b1   = (const float*)d_in[14];
  const float* w2   = (const float*)d_in[15];
  const float* b2   = (const float*)d_in[16];
  float* out = (float*)d_out;

  // Workspace carve-up (~112 MB total).
  char* ws = (char*)d_ws;
  size_t off = 0;
  auto carve = [&](size_t bytes) -> void* {
    void* p = ws + off;
    off += (bytes + 255) & ~(size_t)255;
    return p;
  };
  bf16_t* wq_t   = (bf16_t*)carve((size_t)DD * DD * 2);      // [D x D]
  bf16_t* wk_t   = (bf16_t*)carve((size_t)DD * DD * 2);
  bf16_t* wv_t   = (bf16_t*)carve((size_t)DD * DD * 2);
  bf16_t* wo_t   = (bf16_t*)carve((size_t)DD * DD * 2);
  bf16_t* w1_t   = (bf16_t*)carve((size_t)MLPD * DD * 2);    // [MLP x D]
  bf16_t* w2_t   = (bf16_t*)carve((size_t)DD * MLPD * 2);    // [D x MLP]
  bf16_t* h_bf   = (bf16_t*)carve((size_t)BB * SS * DD * 2); // LN out (reused)
  bf16_t* q_bf   = (bf16_t*)carve((size_t)BB * SS * DD * 2); // [B,H,S,HD]
  bf16_t* k_bf   = (bf16_t*)carve((size_t)BB * SS * DD * 2); // [B,H,S,HD]
  bf16_t* vt_bf  = (bf16_t*)carve((size_t)BB * SS * DD * 2); // [B,H,HD,S]
  bf16_t* at_bf  = (bf16_t*)carve((size_t)BB * SS * DD * 2); // attn out [B,S,D]
  float*  x1     = (float*) carve((size_t)BB * SS * DD * 4); // residual 1
  bf16_t* mlp_bf = (bf16_t*)carve((size_t)BB * SS * MLPD * 2);

  const int ROWS = BB * SS;  // 4096

  auto cvt = [&](const float* w, bf16_t* wt, int K, int N) {
    int total = K * N;
    cvt_kernel<<<dim3((total + 255) / 256), dim3(256), 0, stream>>>(w, wt, K, N);
  };
  auto gemm = [&](const bf16_t* A, const bf16_t* Bt, const float* bias,
                  const float* resid, float* oF, bf16_t* oB, int M, int N,
                  int K, int mode) {
    gemm_bf16_kernel<<<dim3(N / 64, M / 256), dim3(256), 0, stream>>>(
        A, Bt, bias, resid, oF, oB, M, N, K, mode);
  };

  // 1) weights -> bf16, transposed [N x K]
  cvt(wq, wq_t, DD, DD);
  cvt(wk, wk_t, DD, DD);
  cvt(wv, wv_t, DD, DD);
  cvt(wo, wo_t, DD, DD);
  cvt(w1, w1_t, DD, MLPD);
  cvt(w2, w2_t, MLPD, DD);

  // 2) LN1
  ln_kernel<<<dim3(ROWS), dim3(256), 0, stream>>>(x, ln1g, ln1b, h_bf);

  // 3) QKV projections (V written transposed per head)
  gemm(h_bf, wq_t, bq, nullptr, nullptr, q_bf,  ROWS, DD, DD, 0);
  gemm(h_bf, wk_t, bk, nullptr, nullptr, k_bf,  ROWS, DD, DD, 0);
  gemm(h_bf, wv_t, bv, nullptr, nullptr, vt_bf, ROWS, DD, DD, 1);

  // 4) flash attention
  attn_kernel<<<dim3(SS / 64, BB * HH), dim3(128), 0, stream>>>(q_bf, k_bf,
                                                                vt_bf, at_bf);

  // 5) output projection + residual -> x1 (fp32)
  gemm(at_bf, wo_t, bo, x, x1, nullptr, ROWS, DD, DD, 2);

  // 6) LN2 (reuse h_bf)
  ln_kernel<<<dim3(ROWS), dim3(256), 0, stream>>>(x1, ln2g, ln2b, h_bf);

  // 7) MLP up + exact GeLU -> bf16
  gemm(h_bf, w1_t, b1, nullptr, nullptr, mlp_bf, ROWS, MLPD, DD, 3);

  // 8) MLP down + residual -> d_out (fp32)
  gemm(mlp_bf, w2_t, b2, x1, out, nullptr, ROWS, DD, MLPD, 2);
}
